// CGNN_68332929679680
// MI455X (gfx1250) — matrix operand order repaired and checked by reading, not compiled, not run on hardware
//
#include <hip/hip_runtime.h>

typedef __attribute__((ext_vector_type(16))) _Float16 v16h;
typedef __attribute__((ext_vector_type(8)))  _Float16 v8h;
typedef __attribute__((ext_vector_type(8)))  float    v8f;

#define N_NODES  50000
#define N_EDGES  800000
#define HID      64
#define N_LAYERS 3
#define N_GRAPHS 64

// ---------------------------------------------------------------------------
// Pack a 64x64 fp32 weight matrix (row-major, K x N) into WMMA B-fragments.
// Fragment layout (V_WMMA_F32_16X16X32_F16, wave32):
//   lane L: column N = ntile*16 + (L&15); half j -> K = ktile*32 + 8*(L>>4) + j (j<8), +8 more for j>=8
// pk layout: [mat(9)][ktile(2)][ntile(4)][lane(32)][16 halves]
// mats: 0..2 = lin_W[l], 3..5 = mlp_W1[l], 6..8 = mlp_W2[l]
// ---------------------------------------------------------------------------
__global__ void pack_weights(const float* __restrict__ lin_W,
                             const float* __restrict__ mlp_W1,
                             const float* __restrict__ mlp_W2,
                             _Float16* __restrict__ pk) {
  int tid = blockIdx.x * blockDim.x + threadIdx.x;
  if (tid >= 9 * 2 * 4 * 32) return;
  int lane = tid & 31;
  int nt   = (tid >> 5) & 3;
  int kt   = (tid >> 7) & 1;
  int mat  = tid >> 8;
  const float* W;
  if (mat < 3)      W = lin_W  + mat * (HID * HID);
  else if (mat < 6) W = mlp_W1 + (mat - 3) * (HID * HID);
  else              W = mlp_W2 + (mat - 6) * (HID * HID);
  int hi = lane >> 4, lnm = lane & 15;
  int n = nt * 16 + lnm;
  _Float16* out = pk + (size_t)tid * 16;
  for (int j = 0; j < 16; ++j) {
    int k = kt * 32 + hi * 8 + (j < 8 ? j : j + 8);
    out[j] = (_Float16)W[k * HID + n];
  }
}

// h = x @ node_W + node_b    (K=5, scalar; tiny)
__global__ void encode_nodes(const float* __restrict__ x,
                             const float* __restrict__ W,
                             const float* __restrict__ b,
                             float* __restrict__ h) {
  int i = blockIdx.x * blockDim.x + threadIdx.x;
  if (i >= N_NODES * HID) return;
  int node = i >> 6, c = i & 63;
  const float* xr = x + node * 5;
  float acc = b[c];
  #pragma unroll
  for (int k = 0; k < 5; ++k) acc += xr[k] * W[k * HID + c];
  h[i] = acc;
}

// e = edge_attr @ edge_W + edge_b, stored f16 (halves the dominant HBM stream)
__global__ void encode_edges(const float* __restrict__ ea,
                             const float* __restrict__ W,
                             const float* __restrict__ b,
                             _Float16* __restrict__ eh) {
  int i = blockIdx.x * blockDim.x + threadIdx.x;
  if (i >= N_EDGES * HID) return;
  int e = i >> 6, c = i & 63;
  const float* xr = ea + (size_t)e * 5;
  float acc = b[c];
  #pragma unroll
  for (int k = 0; k < 5; ++k) acc += xr[k] * W[k * HID + c];
  eh[i] = (_Float16)acc;
}

__global__ void zero_f32(float* __restrict__ p, int n) {
  int i = blockIdx.x * blockDim.x + threadIdx.x;
  if (i < n) p[i] = 0.0f;
}

// ---------------------------------------------------------------------------
// Edge kernel: one wave = 16 edges.
//   t = e_tile @ lin_W[l] + lin_b[l]   (2 K-tiles x 4 N-tiles of WMMA)
//   m = relu(t + h[src]); atomicAdd agg[dst] += m
// Gathers of h[src] are issued BEFORE the WMMAs so their latency overlaps the
// matrix ops; atomics are batched at the end. Grid is an exact multiple of
// waves -> EXEC all ones at every WMMA.
// ---------------------------------------------------------------------------
__global__ void edge_conv(const _Float16* __restrict__ eh,
                          const _Float16* __restrict__ pk,   // lin_W[l] frags
                          const float* __restrict__ lin_b,   // offset by l*64
                          const float* __restrict__ h,
                          const int* __restrict__ src,
                          const int* __restrict__ dst,
                          float* __restrict__ agg) {
  int wave = (blockIdx.x * blockDim.x + threadIdx.x) >> 5;
  int lane = threadIdx.x & 31;
  int hi = lane >> 4, lnm = lane & 15;
  int base = wave * 16;

  // Rows this lane owns in C/D layout: M = r + 8*hi
  int4 s4a = *(const int4*)(src + base + hi * 8);
  int4 s4b = *(const int4*)(src + base + hi * 8 + 4);
  int4 d4a = *(const int4*)(dst + base + hi * 8);
  int4 d4b = *(const int4*)(dst + base + hi * 8 + 4);
  int s8[8] = {s4a.x, s4a.y, s4a.z, s4a.w, s4b.x, s4b.y, s4b.z, s4b.w};
  int d8[8] = {d4a.x, d4a.y, d4a.z, d4a.w, d4b.x, d4b.y, d4b.z, d4b.w};

  // Issue all 32 gathers of h[src] up-front (independent of the WMMAs).
  float hv[4][8];
  #pragma unroll
  for (int nt = 0; nt < 4; ++nt) {
    int n = nt * 16 + lnm;
    #pragma unroll
    for (int r = 0; r < 8; ++r)
      hv[nt][r] = h[(size_t)s8[r] * HID + n];
  }

  // A fragments (16-bit A layout): halves 0-7 -> K=kt*32+hi*8+j, 8-15 -> +16
  const _Float16* pe = eh + (size_t)(base + lnm) * HID;
  v16h a0, a1;
  {
    v8h lo = *(const v8h*)(pe + hi * 8);
    v8h up = *(const v8h*)(pe + 16 + hi * 8);
    #pragma unroll
    for (int j = 0; j < 8; ++j) { a0[j] = lo[j]; a0[8 + j] = up[j]; }
  }
  {
    v8h lo = *(const v8h*)(pe + 32 + hi * 8);
    v8h up = *(const v8h*)(pe + 48 + hi * 8);
    #pragma unroll
    for (int j = 0; j < 8; ++j) { a1[j] = lo[j]; a1[8 + j] = up[j]; }
  }

  v8f acc[4];
  #pragma unroll
  for (int nt = 0; nt < 4; ++nt) {
    float bias = lin_b[nt * 16 + lnm];
    #pragma unroll
    for (int r = 0; r < 8; ++r) acc[nt][r] = bias;
    v16h b0 = *(const v16h*)(pk + ((size_t)(0 * 4 + nt) * 32 + lane) * 16);
    acc[nt] = __builtin_amdgcn_wmma_f32_16x16x32_f16(false, a0, false, b0, (short)0, acc[nt], false, false);
    v16h b1 = *(const v16h*)(pk + ((size_t)(1 * 4 + nt) * 32 + lane) * 16);
    acc[nt] = __builtin_amdgcn_wmma_f32_16x16x32_f16(false, a1, false, b1, (short)0, acc[nt], false, false);
  }

  // Combine + batched scatter atomics.
  #pragma unroll
  for (int nt = 0; nt < 4; ++nt) {
    int n = nt * 16 + lnm;
    #pragma unroll
    for (int r = 0; r < 8; ++r) {
      float v = acc[nt][r] + hv[nt][r];
      v = v > 0.0f ? v : 0.0f;
      atomicAdd(&agg[(size_t)d8[r] * HID + n], v);
    }
  }
}

// ---------------------------------------------------------------------------
// Node kernel: one wave = 16 nodes.
//   z  = h + agg
//   y1 = relu(z @ mlp_W1 + b1)           (WMMA, via LDS transpose)
//   y2 = y1 @ mlp_W2 + b2
//   h  = relu(bn_g * y2 * inv_std + bn_b)   (coalesced f4 write-back via LDS)
// ---------------------------------------------------------------------------
__global__ void node_mlp(float* __restrict__ h,
                         const float* __restrict__ agg,
                         const _Float16* __restrict__ pk1,
                         const _Float16* __restrict__ pk2,
                         const float* __restrict__ b1,
                         const float* __restrict__ b2,
                         const float* __restrict__ bng,
                         const float* __restrict__ bnb,
                         int ntiles) {
  __shared__ __align__(16) float lds[8 * 16 * HID];  // per-wave 16x64 tile
  int wv   = threadIdx.x >> 5;
  int tile = blockIdx.x * 8 + wv;
  bool valid = tile < ntiles;                   // wave-uniform guard
  int lane = threadIdx.x & 31;
  int hi = lane >> 4, lnm = lane & 15;
  int base = tile * 16;
  float* myl = lds + wv * (16 * HID);

  if (valid) {
    const float* ph = h   + (size_t)(base + lnm) * HID;
    const float* pa = agg + (size_t)(base + lnm) * HID;
    v16h a0, a1;
    #pragma unroll
    for (int t = 0; t < 2; ++t) {
      int k0 = t * 32 + hi * 8;
      v16h a;
      #pragma unroll
      for (int j = 0; j < 8; ++j) {
        a[j]     = (_Float16)(ph[k0 + j]      + pa[k0 + j]);
        a[8 + j] = (_Float16)(ph[k0 + 16 + j] + pa[k0 + 16 + j]);
      }
      if (t == 0) a0 = a; else a1 = a;
    }
    #pragma unroll
    for (int nt = 0; nt < 4; ++nt) {
      int n = nt * 16 + lnm;
      float bias = b1[n];
      v8f acc;
      #pragma unroll
      for (int r = 0; r < 8; ++r) acc[r] = bias;
      v16h w0 = *(const v16h*)(pk1 + ((size_t)(0 * 4 + nt) * 32 + lane) * 16);
      acc = __builtin_amdgcn_wmma_f32_16x16x32_f16(false, a0, false, w0, (short)0, acc, false, false);
      v16h w1 = *(const v16h*)(pk1 + ((size_t)(1 * 4 + nt) * 32 + lane) * 16);
      acc = __builtin_amdgcn_wmma_f32_16x16x32_f16(false, a1, false, w1, (short)0, acc, false, false);
      #pragma unroll
      for (int r = 0; r < 8; ++r) {
        float v = acc[r];
        myl[(r + 8 * hi) * HID + n] = v > 0.0f ? v : 0.0f;
      }
    }
  }
  __syncthreads();
  if (valid) {
    const float* pl = myl + lnm * HID;          // row = lane's M
    v16h a0, a1;
    #pragma unroll
    for (int t = 0; t < 2; ++t) {
      int k0 = t * 32 + hi * 8;
      v16h a;
      #pragma unroll
      for (int j = 0; j < 8; ++j) {
        a[j]     = (_Float16)pl[k0 + j];
        a[8 + j] = (_Float16)pl[k0 + 16 + j];
      }
      if (t == 0) a0 = a; else a1 = a;
    }
    const float inv_std = 0.9999950000374997f;  // 1/sqrt(1 + 1e-5)
    #pragma unroll
    for (int nt = 0; nt < 4; ++nt) {
      int n = nt * 16 + lnm;
      float bias = b2[n];
      v8f acc;
      #pragma unroll
      for (int r = 0; r < 8; ++r) acc[r] = bias;
      v16h w0 = *(const v16h*)(pk2 + ((size_t)(0 * 4 + nt) * 32 + lane) * 16);
      acc = __builtin_amdgcn_wmma_f32_16x16x32_f16(false, a0, false, w0, (short)0, acc, false, false);
      v16h w1 = *(const v16h*)(pk2 + ((size_t)(1 * 4 + nt) * 32 + lane) * 16);
      acc = __builtin_amdgcn_wmma_f32_16x16x32_f16(false, a1, false, w1, (short)0, acc, false, false);
      float g = bng[n], bb = bnb[n];
      #pragma unroll
      for (int r = 0; r < 8; ++r) {
        float v = g * (acc[r] * inv_std) + bb;
        myl[(r + 8 * hi) * HID + n] = v > 0.0f ? v : 0.0f;
      }
    }
  }
  __syncthreads();
  if (valid) {
    // Coalesced write-back: 8 x (32 lanes x float4) covers the 16x64 tile.
    float* hb = h + (size_t)base * HID;
    #pragma unroll
    for (int j = 0; j < 8; ++j) {
      int idx = j * 128 + lane * 4;
      *(float4*)(hb + idx) = *(const float4*)(myl + idx);
    }
  }
}

// ---------------------------------------------------------------------------
// Graph pooling: sum / count via f32 atomics, max via int-bit atomicMax
// (valid since h >= 0 post-ReLU; init 0 matches the counts==0 mask).
// ---------------------------------------------------------------------------
__global__ void pool_graphs(const float* __restrict__ h,
                            const int* __restrict__ batch,
                            float* __restrict__ hsum,
                            float* __restrict__ hmax,
                            float* __restrict__ counts) {
  int i = blockIdx.x * blockDim.x + threadIdx.x;
  if (i >= N_NODES * HID) return;
  int node = i >> 6, c = i & 63;
  int g = batch[node];
  float v = h[i];
  atomicAdd(&hsum[g * HID + c], v);
  atomicMax((int*)&hmax[g * HID + c], __float_as_int(v));
  if (c == 0) atomicAdd(&counts[g], 1.0f);
}

// Head: relu([mean|max|sum] @ W1 + b1) @ W2 + b2 -> [64,5]
__global__ void head_mlp(const float* __restrict__ hsum,
                         const float* __restrict__ hmax,
                         const float* __restrict__ counts,
                         const float* __restrict__ W1,
                         const float* __restrict__ b1,
                         const float* __restrict__ W2,
                         const float* __restrict__ b2,
                         float* __restrict__ out) {
  __shared__ float hc[3 * HID];
  __shared__ float y1[HID];
  int g = blockIdx.x, t = threadIdx.x;    // 64 threads
  float cnt = counts[g];
  float s = hsum[g * HID + t];
  hc[t]           = s / fmaxf(cnt, 1.0f);
  hc[HID + t]     = cnt > 0.0f ? hmax[g * HID + t] : 0.0f;
  hc[2 * HID + t] = s;
  __syncthreads();
  float acc = b1[t];
  for (int k = 0; k < 3 * HID; ++k) acc += hc[k] * W1[k * HID + t];
  y1[t] = acc > 0.0f ? acc : 0.0f;
  __syncthreads();
  if (t < 5) {
    float a2 = b2[t];
    for (int k = 0; k < HID; ++k) a2 += y1[k] * W2[k * 5 + t];
    out[g * 5 + t] = a2;
  }
}

extern "C" void kernel_launch(void* const* d_in, const int* in_sizes, int n_in,
                              void* d_out, int out_size, void* d_ws, size_t ws_size,
                              hipStream_t stream) {
  const float* x         = (const float*)d_in[0];
  const float* edge_attr = (const float*)d_in[1];
  const int*   edge_idx  = (const int*)d_in[2];
  const int*   batch     = (const int*)d_in[3];
  const float* node_W    = (const float*)d_in[4];
  const float* node_b    = (const float*)d_in[5];
  const float* edge_W    = (const float*)d_in[6];
  const float* edge_b    = (const float*)d_in[7];
  const float* lin_W     = (const float*)d_in[8];
  const float* lin_b     = (const float*)d_in[9];
  const float* mlp_W1    = (const float*)d_in[10];
  const float* mlp_b1    = (const float*)d_in[11];
  const float* mlp_W2    = (const float*)d_in[12];
  const float* mlp_b2    = (const float*)d_in[13];
  const float* bn_g      = (const float*)d_in[14];
  const float* bn_b      = (const float*)d_in[15];
  const float* head_W1   = (const float*)d_in[16];
  const float* head_b1   = (const float*)d_in[17];
  const float* head_W2   = (const float*)d_in[18];
  const float* head_b2   = (const float*)d_in[19];
  float* out = (float*)d_out;

  // Workspace carve-up (256B aligned blocks)
  char* ws = (char*)d_ws;
  size_t off = 0;
  auto take = [&](size_t bytes) {
    char* p = ws + off;
    off += (bytes + 255) & ~(size_t)255;
    return p;
  };
  float*    h      = (float*)take((size_t)N_NODES * HID * 4);
  float*    agg    = (float*)take((size_t)N_NODES * HID * 4);
  _Float16* eh     = (_Float16*)take((size_t)N_EDGES * HID * 2);
  _Float16* pk     = (_Float16*)take((size_t)9 * 2 * 4 * 32 * 16 * 2);
  float*    hsum   = (float*)take((size_t)N_GRAPHS * HID * 4);
  float*    hmax   = (float*)take((size_t)N_GRAPHS * HID * 4);
  float*    counts = (float*)take((size_t)N_GRAPHS * 4);

  const int* src = edge_idx;
  const int* dst = edge_idx + N_EDGES;

  pack_weights<<<9, 256, 0, stream>>>(lin_W, mlp_W1, mlp_W2, pk);
  encode_nodes<<<(N_NODES * HID) / 256, 256, 0, stream>>>(x, node_W, node_b, h);
  encode_edges<<<(N_EDGES * HID) / 256, 256, 0, stream>>>(edge_attr, edge_W, edge_b, eh);

  const int node_tiles = N_NODES / 16;           // 3125
  const int node_blocks = (node_tiles + 7) / 8;  // 391
  for (int l = 0; l < N_LAYERS; ++l) {
    zero_f32<<<(N_NODES * HID) / 256, 256, 0, stream>>>(agg, N_NODES * HID);
    edge_conv<<<(N_EDGES / 16) / 8, 256, 0, stream>>>(
        eh, pk + (size_t)l * 4096, lin_b + l * HID, h, src, dst, agg);
    node_mlp<<<node_blocks, 256, 0, stream>>>(
        h, agg, pk + (size_t)(3 + l) * 4096, pk + (size_t)(6 + l) * 4096,
        mlp_b1 + l * HID, mlp_b2 + l * HID, bn_g + l * HID, bn_b + l * HID,
        node_tiles);
  }

  // pool buffers are contiguous in ws: hsum(4096f) hmax(4096f) counts(64f)
  zero_f32<<<33, 256, 0, stream>>>(hsum, 2 * N_GRAPHS * HID + N_GRAPHS);
  pool_graphs<<<(N_NODES * HID) / 256, 256, 0, stream>>>(h, batch, hsum, hmax, counts);
  head_mlp<<<N_GRAPHS, 64, 0, stream>>>(hsum, hmax, counts, head_W1, head_b1,
                                        head_W2, head_b2, out);
}
